// BSplineBasis_1606317769442
// MI455X (gfx1250) — compile-verified, matrix operand order checked by reading, not run on hardware
//
#include <hip/hip_runtime.h>

// KAN layer as one bf16-WMMA GEMM:  out = [basis | x] @ [coeffs ; W]
// Path A (ws >= 4.7MB): prepack B (coeffs+base_weight) to bf16 chunk tiles in
//   d_ws once; main kernel loads B fragments straight from L2 into VGPRs
//   (register ping-pong, one chunk ahead), double-buffers the LDS A tile
//   (on-the-fly spline basis), 1 barrier/chunk, 2x-unrolled chunk loop.
// Path B (fallback): fully fused kernel converting B in-kernel.

#define IN_F  512
#define OUT_F 512
#define NCTRL 8
#define BM    64     // rows of A per workgroup
#define KC    32     // K per chunk
#define ASTR  40     // padded LDS row stride in u16 (80 B)
#define BSTR  40
#define NSPLINE (IN_F / 4)            // 128 spline chunks (4 features x 8 ctrl)
#define NCHUNK  (NSPLINE + IN_F / KC) // + 16 residual chunks = 144 (even)

typedef __attribute__((ext_vector_type(16))) __bf16 v16bf;
typedef __attribute__((ext_vector_type(2)))  __bf16 v2bf;
typedef __attribute__((ext_vector_type(8)))  float  v8f;
typedef unsigned short u16;
typedef unsigned int   u32;

struct Frag32B { uint4 lo, hi; };

__device__ __forceinline__ u32 pack2(float a, float b) {
#if __has_builtin(__builtin_amdgcn_cvt_pk_bf16_f32)
  v2bf v = __builtin_amdgcn_cvt_pk_bf16_f32(a, b);
  return __builtin_bit_cast(u32, v);
#else
  v2bf v = { (__bf16)a, (__bf16)b };
  return __builtin_bit_cast(u32, v);
#endif
}

__device__ __forceinline__ u16 f2bf(float f) {
  u32 u = __builtin_bit_cast(u32, f);
  return (u16)((u + 0x7fffu + ((u >> 16) & 1u)) >> 16);
}

// WMMA 16x16x32 bf16 fragment (ISA 7.12.2): lane = M/N = lane&15, half=lane>>4.
// Elements 0..7: K = half*8 + e ; elements 8..15: K = 16 + half*8 + (e-8)
// -> two contiguous 16-byte runs at k offsets half*8 and 16+half*8.
__device__ __forceinline__ v16bf load_frag(const u16* row, int half) {
  Frag32B f;
  f.lo = *(const uint4*)(row + half * 8);
  f.hi = *(const uint4*)(row + 16 + half * 8);
  return __builtin_bit_cast(v16bf, f);
}
__device__ __forceinline__ v16bf load_bfrag(const u16* base) {  // base includes half*8
  Frag32B f;
  f.lo = *(const uint4*)(base);
  f.hi = *(const uint4*)(base + 16);
  return __builtin_bit_cast(v16bf, f);
}

// Cubic B-spline basis, uniform grid g_j = -2.2 + 0.4 j (j=0..11), 8 outputs.
__device__ __forceinline__ void bspline8(float t, float* o) {
  const float h = 0.4f;
  float b0[11];
#pragma unroll
  for (int j = 0; j < 11; ++j) {
    float gj  = -2.2f + h * (float)j;
    float gj1 = -2.2f + h * (float)(j + 1);
    b0[j] = (t >= gj && t < gj1) ? 1.0f : 0.0f;
  }
  float b1[10];
#pragma unroll
  for (int j = 0; j < 10; ++j) {
    float gj  = -2.2f + h * (float)j;
    float gj2 = -2.2f + h * (float)(j + 2);
    b1[j] = (t - gj) * 2.5f * b0[j] + (gj2 - t) * 2.5f * b0[j + 1];
  }
  float b2[9];
#pragma unroll
  for (int j = 0; j < 9; ++j) {
    float gj  = -2.2f + h * (float)j;
    float gj3 = -2.2f + h * (float)(j + 3);
    b2[j] = (t - gj) * 1.25f * b1[j] + (gj3 - t) * 1.25f * b1[j + 1];
  }
#pragma unroll
  for (int j = 0; j < 8; ++j) {
    float gj  = -2.2f + h * (float)j;
    float gj4 = -2.2f + h * (float)(j + 4);
    o[j] = (t - gj) * 0.83333333f * b2[j] + (gj4 - t) * 0.83333333f * b2[j + 1];
  }
}

// ---------------------------------------------------------------------------
// Prepack: B matrix (coeffs flattened over (i,n), then base_weight) -> bf16
// chunk tiles wsB[c][o][kk], c<144, o<512, kk<32. One thread per (c,o).
// ---------------------------------------------------------------------------
__global__ __launch_bounds__(256) void prepack_b_kernel(
    const float* __restrict__ coeffs, const float* __restrict__ bw,
    u16* __restrict__ wsB) {
  const int p = blockIdx.x * 256 + threadIdx.x;   // 144*512 tasks
  const int c = p >> 9, o = p & (OUT_F - 1);
  u16* dst = wsB + (size_t)p * KC;
  if (c < NSPLINE) {
    const int f0 = c * 4;
#pragma unroll
    for (int fi = 0; fi < 4; ++fi) {
      const float* src = coeffs + ((size_t)(f0 + fi) * OUT_F + o) * NCTRL;
      float4 c0 = *(const float4*)(src);
      float4 c1 = *(const float4*)(src + 4);
      uint4 v;
      v.x = pack2(c0.x, c0.y); v.y = pack2(c0.z, c0.w);
      v.z = pack2(c1.x, c1.y); v.w = pack2(c1.z, c1.w);
      *(uint4*)(dst + fi * 8) = v;
    }
  } else {
    const int k0 = (c - NSPLINE) * KC;
#pragma unroll
    for (int q = 0; q < 4; ++q) {
      float a[8];
#pragma unroll
      for (int j = 0; j < 8; ++j) a[j] = bw[(size_t)(k0 + q * 8 + j) * OUT_F + o];
      uint4 v;
      v.x = pack2(a[0], a[1]); v.y = pack2(a[2], a[3]);
      v.z = pack2(a[4], a[5]); v.w = pack2(a[6], a[7]);
      *(uint4*)(dst + q * 8) = v;
    }
  }
}

// ---------------------------------------------------------------------------
// Main kernel (prepacked-B path): software-pipelined, double-buffered A tile,
// 2x-unrolled chunk loop with ping-pong B fragment sets (no register copies).
// ---------------------------------------------------------------------------
__global__ __launch_bounds__(256) void kan_wmma_ws_kernel(
    const float* __restrict__ x, const u16* __restrict__ wsB,
    float* __restrict__ out, int B) {
  __shared__ u16 At[2][BM * ASTR];    // double-buffered 64x32 bf16 A chunk

  const int tid  = threadIdx.x;
  const int w    = tid >> 5;          // wave 0..7 -> cols w*64 .. +63
  const int lane = tid & 31;
  const int half = lane >> 4, ln = lane & 15;
  const int rowBlock = blockIdx.x * BM;

  if (blockIdx.x == 0 && tid == 0) out[(size_t)B * OUT_F] = 0.0f;  // kl

  v8f acc[4][4] = {};

  auto fillA = [&](int c, int buf) {
    if (c < NSPLINE) {
      const int f0 = c * 4;
      const int row = tid >> 2, fi = tid & 3;
      float t = tanhf(x[(size_t)(rowBlock + row) * IN_F + f0 + fi]);
      float b[8];
      bspline8(t, b);
      uint4 v;
      v.x = pack2(b[0], b[1]); v.y = pack2(b[2], b[3]);
      v.z = pack2(b[4], b[5]); v.w = pack2(b[6], b[7]);
      *(uint4*)&At[buf][row * ASTR + fi * 8] = v;
    } else {
      const int k0 = (c - NSPLINE) * KC;
#pragma unroll
      for (int s = 0; s < 2; ++s) {
        int p = s * 256 + tid;
        int row = p >> 3, kq = p & 7;
        float4 v = *(const float4*)&x[(size_t)(rowBlock + row) * IN_F + k0 + kq * 4];
        uint2 u; u.x = pack2(v.x, v.y); u.y = pack2(v.z, v.w);
        *(uint2*)&At[buf][row * ASTR + kq * 4] = u;
      }
    }
  };
  auto bfragPtr = [&](int c, int ct) -> const u16* {
    int col = w * 64 + ct * 16 + ln;
    return wsB + ((size_t)(c * OUT_F + col) * KC + half * 8);
  };
  auto consume = [&](const u16* Ac, v16bf (&bf)[4]) {
#pragma unroll
    for (int mt = 0; mt < 4; ++mt) {
      v16bf af = load_frag(Ac + (mt * 16 + ln) * ASTR, half);
#pragma unroll
      for (int ct = 0; ct < 4; ++ct)
        acc[mt][ct] = __builtin_amdgcn_wmma_f32_16x16x32_bf16(
            false, af, false, bf[ct], (short)0, acc[mt][ct], false, false);
    }
  };

  v16bf bA[4], bB[4];
  fillA(0, 0);
#pragma unroll
  for (int ct = 0; ct < 4; ++ct) bA[ct] = load_bfrag(bfragPtr(0, ct));
  __syncthreads();

  for (int c = 0; c < NCHUNK; c += 2) {
    // phase 0: consume chunk c (buffer 0, frag set A); prefetch c+1 into set B
#pragma unroll
    for (int ct = 0; ct < 4; ++ct) bB[ct] = load_bfrag(bfragPtr(c + 1, ct));
    fillA(c + 1, 1);
    consume(At[0], bA);
    __syncthreads();

    // phase 1: consume chunk c+1 (buffer 1, frag set B); prefetch c+2 into set A
    if (c + 2 < NCHUNK) {
#pragma unroll
      for (int ct = 0; ct < 4; ++ct) bA[ct] = load_bfrag(bfragPtr(c + 2, ct));
      fillA(c + 2, 0);
    }
    consume(At[1], bB);
    __syncthreads();
  }

#pragma unroll
  for (int mt = 0; mt < 4; ++mt)
#pragma unroll
    for (int ct = 0; ct < 4; ++ct) {
      int gc = w * 64 + ct * 16 + ln;
#pragma unroll
      for (int v = 0; v < 8; ++v)
        out[(size_t)(rowBlock + mt * 16 + half * 8 + v) * OUT_F + gc] = acc[mt][ct][v];
    }
}

// ---------------------------------------------------------------------------
// Fallback (no workspace): fully fused kernel.
// ---------------------------------------------------------------------------
__global__ __launch_bounds__(256) void kan_wmma_kernel(
    const float* __restrict__ x, const float* __restrict__ coeffs,
    const float* __restrict__ bw, float* __restrict__ out, int B) {
  __shared__ u16 At[BM * ASTR];
  __shared__ u16 Bt[OUT_F * BSTR];

  const int tid  = threadIdx.x;
  const int w    = tid >> 5;
  const int lane = tid & 31;
  const int half = lane >> 4, ln = lane & 15;
  const int rowBlock = blockIdx.x * BM;

  if (blockIdx.x == 0 && tid == 0) out[(size_t)B * OUT_F] = 0.0f;

  v8f acc[4][4] = {};

  for (int c = 0; c < NCHUNK; ++c) {
    __syncthreads();
    if (c < NSPLINE) {
      const int f0 = c * 4;
      {
        const int row = tid >> 2, fi = tid & 3;
        float t = tanhf(x[(size_t)(rowBlock + row) * IN_F + f0 + fi]);
        float b[8];
        bspline8(t, b);
        uint4 v;
        v.x = pack2(b[0], b[1]); v.y = pack2(b[2], b[3]);
        v.z = pack2(b[4], b[5]); v.w = pack2(b[6], b[7]);
        *(uint4*)&At[row * ASTR + fi * 8] = v;
      }
#pragma unroll
      for (int s = 0; s < 8; ++s) {
        int p  = s * 256 + tid;
        int fi = p >> 9, o = p & (OUT_F - 1);
        const float* src = coeffs + ((size_t)(f0 + fi) * OUT_F + o) * NCTRL;
        float4 c0 = *(const float4*)(src);
        float4 c1 = *(const float4*)(src + 4);
        uint4 v;
        v.x = pack2(c0.x, c0.y); v.y = pack2(c0.z, c0.w);
        v.z = pack2(c1.x, c1.y); v.w = pack2(c1.z, c1.w);
        *(uint4*)&Bt[o * BSTR + fi * 8] = v;
      }
    } else {
      const int k0 = (c - NSPLINE) * KC;
#pragma unroll
      for (int s = 0; s < 2; ++s) {
        int p = s * 256 + tid;
        int row = p >> 3, kq = p & 7;
        float4 v = *(const float4*)&x[(size_t)(rowBlock + row) * IN_F + k0 + kq * 4];
        uint2 u; u.x = pack2(v.x, v.y); u.y = pack2(v.z, v.w);
        *(uint2*)&At[row * ASTR + kq * 4] = u;
      }
#pragma unroll
      for (int s = 0; s < 16; ++s) {
        int p = s * 256 + tid;
        int kk = p >> 7, cq = p & 127;
        float4 v = *(const float4*)&bw[(size_t)(k0 + kk) * OUT_F + cq * 4];
        Bt[(cq * 4 + 0) * BSTR + kk] = f2bf(v.x);
        Bt[(cq * 4 + 1) * BSTR + kk] = f2bf(v.y);
        Bt[(cq * 4 + 2) * BSTR + kk] = f2bf(v.z);
        Bt[(cq * 4 + 3) * BSTR + kk] = f2bf(v.w);
      }
    }
    __syncthreads();

    v16bf bf[4];
#pragma unroll
    for (int ct = 0; ct < 4; ++ct)
      bf[ct] = load_frag(&Bt[(w * 64 + ct * 16 + ln) * BSTR], half);
#pragma unroll
    for (int mt = 0; mt < 4; ++mt) {
      v16bf af = load_frag(&At[(mt * 16 + ln) * ASTR], half);
#pragma unroll
      for (int ct = 0; ct < 4; ++ct)
        acc[mt][ct] = __builtin_amdgcn_wmma_f32_16x16x32_bf16(
            false, af, false, bf[ct], (short)0, acc[mt][ct], false, false);
    }
  }

#pragma unroll
  for (int mt = 0; mt < 4; ++mt)
#pragma unroll
    for (int ct = 0; ct < 4; ++ct) {
      int gc = w * 64 + ct * 16 + ln;
#pragma unroll
      for (int v = 0; v < 8; ++v)
        out[(size_t)(rowBlock + mt * 16 + half * 8 + v) * OUT_F + gc] = acc[mt][ct][v];
    }
}

extern "C" void kernel_launch(void* const* d_in, const int* in_sizes, int n_in,
                              void* d_out, int out_size, void* d_ws, size_t ws_size,
                              hipStream_t stream) {
  (void)out_size; (void)n_in;
  const float* x      = (const float*)d_in[0];
  const float* coeffs = (const float*)d_in[1];
  const float* bw     = (const float*)d_in[2];
  float* out          = (float*)d_out;
  const int B = in_sizes[0] / IN_F;   // 16384

  const size_t needB = (size_t)NCHUNK * OUT_F * KC * sizeof(u16);  // 4.72 MB
  if (d_ws && ws_size >= needB) {
    u16* wsB = (u16*)d_ws;
    hipLaunchKernelGGL(prepack_b_kernel, dim3(NCHUNK * OUT_F / 256), dim3(256),
                       0, stream, coeffs, bw, wsB);
    hipLaunchKernelGGL(kan_wmma_ws_kernel, dim3(B / BM), dim3(256),
                       0, stream, x, wsB, out, B);
  } else {
    hipLaunchKernelGGL(kan_wmma_kernel, dim3(B / BM), dim3(256),
                       0, stream, x, coeffs, bw, out, B);
  }
}
